// CausalSelfAttention_14242111553971
// MI455X (gfx1250) — matrix operand order repaired and checked by reading, not compiled, not run on hardware
//
#include <hip/hip_runtime.h>

// ---------------------------------------------------------------------------
// CDNA5 (gfx1250) causal self-attention, bf16 WMMA pipeline.
//   x[B,T,E] -> qkv = x@Wqkv + b  -> flash attention -> y@Wout + b -> out
//   B=4, T=2048, E=1024, H=16, hd=64
// ---------------------------------------------------------------------------

typedef __attribute__((ext_vector_type(16))) __bf16 v16bf;
typedef __attribute__((ext_vector_type(8)))  float  v8f;
typedef __attribute__((__vector_size__(4 * sizeof(int)))) int v4i;

#define DEV __device__ __forceinline__

#if defined(__has_builtin)
# if __has_builtin(__builtin_amdgcn_global_load_async_to_lds_b128)
#  define HAVE_ASYNC 1
# endif
#endif
#ifndef HAVE_ASYNC
# define HAVE_ASYNC 0
#endif

typedef __attribute__((address_space(1))) v4i* gptr_v4i;
typedef __attribute__((address_space(3))) v4i* lptr_v4i;

DEV unsigned short f2bf(float f) {
    unsigned u = __float_as_uint(f);
    unsigned r = u + 0x7FFFu + ((u >> 16) & 1u);   // round-to-nearest-even
    return (unsigned short)(r >> 16);
}

DEV v8f vzero8() { v8f z = {0.f,0.f,0.f,0.f,0.f,0.f,0.f,0.f}; return z; }

union FragU { v16bf v; unsigned u[8]; unsigned short h[16]; };
union Pack16 { uint4 v[2]; unsigned short h[16]; };

// Copy 32 contiguous bf16 (64B) global -> LDS.
// Async (GLOBAL_LOAD_ASYNC_TO_LDS_B128, ASYNCcnt) when available.
DEV void copy64B_g2l(const unsigned short* __restrict__ g,
                     unsigned short* __restrict__ l) {
#if HAVE_ASYNC
    __builtin_amdgcn_global_load_async_to_lds_b128(
        (gptr_v4i)(void*)(g),      (lptr_v4i)(void*)(l),      0, 0);
    __builtin_amdgcn_global_load_async_to_lds_b128(
        (gptr_v4i)(void*)(g + 8),  (lptr_v4i)(void*)(l + 8),  0, 0);
    __builtin_amdgcn_global_load_async_to_lds_b128(
        (gptr_v4i)(void*)(g + 16), (lptr_v4i)(void*)(l + 16), 0, 0);
    __builtin_amdgcn_global_load_async_to_lds_b128(
        (gptr_v4i)(void*)(g + 24), (lptr_v4i)(void*)(l + 24), 0, 0);
#else
    ((uint4*)l)[0] = ((const uint4*)g)[0];
    ((uint4*)l)[1] = ((const uint4*)g)[1];
#endif
}

DEV void wait_async() {
#if HAVE_ASYNC
# if __has_builtin(__builtin_amdgcn_s_wait_asynccnt)
    __builtin_amdgcn_s_wait_asynccnt(0);
# else
    asm volatile("s_wait_asynccnt 0" ::: "memory");
# endif
#endif
}

// A-matrix fragment (16x32 bf16, M x K). src = base of 16-row tile, row stride
// `ld` (elements). lane m = lane&15; half = lane>>4;
//   VGPR j   (j<4):  K = half*8  + 2j, 2j+1
//   VGPR 4+j (j<4):  K = 16 + half*8 + 2j, 2j+1
DEV v16bf load_frag_A(const unsigned short* __restrict__ src, int ld) {
    const int lane = threadIdx.x & 31;
    const int m = lane & 15, half = lane >> 4;
    const unsigned* row = (const unsigned*)(src + (size_t)m * ld);
    FragU r;
#pragma unroll
    for (int j = 0; j < 4; ++j) r.u[j]     = row[half * 4 + j];
#pragma unroll
    for (int j = 0; j < 4; ++j) r.u[4 + j] = row[8 + half * 4 + j];
    return r.v;
}

// B-matrix fragment (32x16 bf16, K x N) from K-minor ("transposed") storage:
// element B[kk][n] lives at src[n*ld + kk], so the (2j,2j+1) K-pair is
// contiguous -> compiler merges the 8 b32 reads into 2x ds_load_b128.
// lane n = lane&15; half = lane>>4; VGPR j (j<4): K = half*8 + 2j, 2j+1;
// VGPR 4+j (j<4): K = 16 + half*8 + 2j, 2j+1.
DEV v16bf load_frag_BT(const unsigned short* __restrict__ src, int ld) {
    const int lane = threadIdx.x & 31;
    const int n = lane & 15, half = lane >> 4;
    const unsigned* row = (const unsigned*)(src + (size_t)n * ld);
    FragU r;
#pragma unroll
    for (int j = 0; j < 4; ++j) r.u[j]     = row[half * 4 + j];
#pragma unroll
    for (int j = 0; j < 4; ++j) r.u[4 + j] = row[8 + half * 4 + j];
    return r.v;
}

DEV float rowmax16(float v) {
#pragma unroll
    for (int m = 1; m <= 8; m <<= 1) v = fmaxf(v, __shfl_xor(v, m, 32));
    return v;
}
DEV float rowsum16(float v) {
#pragma unroll
    for (int m = 1; m <= 8; m <<= 1) v += __shfl_xor(v, m, 32);
    return v;
}

#define WMMA_BF16(A_, B_, C_) \
    __builtin_amdgcn_wmma_f32_16x16x32_bf16(false, (A_), false, (B_), (short)0, (C_), false, false)

// ---------------------------------------------------------------------------
// f32 -> bf16 conversion (vectorized, grid-stride)
// ---------------------------------------------------------------------------
__global__ __launch_bounds__(256) void cvt_f32_bf16(
    const float* __restrict__ in, unsigned short* __restrict__ out, int n4) {
    int i = blockIdx.x * blockDim.x + threadIdx.x;
    const int stride = gridDim.x * blockDim.x;
    const float4* in4 = (const float4*)in;
    ushort4* out4 = (ushort4*)out;
    for (; i < n4; i += stride) {
        float4 v = in4[i];
        ushort4 o;
        o.x = f2bf(v.x); o.y = f2bf(v.y); o.z = f2bf(v.z); o.w = f2bf(v.w);
        out4[i] = o;
    }
}

// ---------------------------------------------------------------------------
// Tiled bf16 GEMM:  C[M,N] = A[M,K] @ B[K,N] + bias,  f32 accumulate.
// BM=128 BN=128 BK=32, 256 threads (8 waves), wave grid 2x4, each wave 64x32.
// Double-buffered LDS; A staged via async-to-LDS, B staged transposed.
// ---------------------------------------------------------------------------
template <bool OUT_BF16>
__global__ __launch_bounds__(256) void gemm_bf16(
    const unsigned short* __restrict__ A,   // [M,K] row-major bf16
    const unsigned short* __restrict__ Bm,  // [K,N] row-major bf16
    const float* __restrict__ bias,         // [N] f32
    void* __restrict__ Cout,                // bf16 or f32 [M,N]
    int M, int N, int K) {
    constexpr int BM = 128, BN = 128, BK = 32;
    constexpr int LDA  = 48;   // 96B row stride (16B aligned)
    constexpr int LDBT = 48;   // Bs is [BN][LDBT], K-minor
    __shared__ unsigned short As[2][BM * LDA];
    __shared__ unsigned short Bs[2][BN * LDBT];

    const int tid  = threadIdx.x;
    const int wave = tid >> 5;
    const int lane = tid & 31;
    const int wm = wave >> 2;          // 0..1
    const int wn = wave & 3;           // 0..3
    const int m0 = blockIdx.y * BM;
    const int n0 = blockIdx.x * BN;
    const int ln = lane & 15, lh = lane >> 4;

    const int ar = tid >> 1, ac = (tid & 1) * 16;       // A tile: 128 x 32
    const int br = tid >> 3, bc = (tid & 7) * 16;       // B tile: 32 x 128

    v8f acc[4][2];
#pragma unroll
    for (int i = 0; i < 4; ++i)
#pragma unroll
        for (int j = 0; j < 2; ++j) acc[i][j] = vzero8();

    // ---- prologue: stage tile 0 ----
    copy64B_g2l(A + (size_t)(m0 + ar) * K + ac, &As[0][ar * LDA + ac]);
    {
        Pack16 p;
        p.v[0] = ((const uint4*)(Bm + (size_t)br * N + n0 + bc))[0];
        p.v[1] = ((const uint4*)(Bm + (size_t)br * N + n0 + bc))[1];
#pragma unroll
        for (int i = 0; i < 16; ++i) Bs[0][(bc + i) * LDBT + br] = p.h[i];
    }
    wait_async();
    __syncthreads();

    int cur = 0;
    for (int k0 = 0; k0 < K; k0 += BK) {
        const bool has_next = (k0 + BK) < K;
        Pack16 nb;
        if (has_next) {
            copy64B_g2l(A + (size_t)(m0 + ar) * K + k0 + BK + ac,
                        &As[cur ^ 1][ar * LDA + ac]);
            const uint4* src = (const uint4*)(Bm + (size_t)(k0 + BK + br) * N + n0 + bc);
            nb.v[0] = src[0];
            nb.v[1] = src[1];
            __builtin_prefetch(A + (size_t)(m0 + ar) * K + k0 + 2 * BK + ac, 0, 1);
        }

        // ---- compute on current buffers: batch ALL fragment loads first so
        // the ds_load_b128s issue back-to-back and overlap with the WMMAs ----
        v16bf bfrag[2], afrag[4];
#pragma unroll
        for (int ni = 0; ni < 2; ++ni)
            bfrag[ni] = load_frag_BT(&Bs[cur][(wn * 32 + ni * 16) * LDBT], LDBT);
#pragma unroll
        for (int mi = 0; mi < 4; ++mi)
            afrag[mi] = load_frag_A(&As[cur][(wm * 64 + mi * 16) * LDA], LDA);
#pragma unroll
        for (int mi = 0; mi < 4; ++mi)
#pragma unroll
            for (int ni = 0; ni < 2; ++ni)
                acc[mi][ni] = WMMA_BF16(afrag[mi], bfrag[ni], acc[mi][ni]);

        if (has_next) {
#pragma unroll
            for (int i = 0; i < 16; ++i)
                Bs[cur ^ 1][(bc + i) * LDBT + br] = nb.h[i];
        }
        wait_async();
        __syncthreads();
        cur ^= 1;
    }

    // store (C layout: lane n = ln, rows m = lh*8 + r in VGPR r)
#pragma unroll
    for (int mi = 0; mi < 4; ++mi) {
#pragma unroll
        for (int ni = 0; ni < 2; ++ni) {
            const int col = n0 + wn * 32 + ni * 16 + ln;
            const float bv = bias[col];
            v8f a = acc[mi][ni];
#pragma unroll
            for (int r = 0; r < 8; ++r) {
                const int row = m0 + wm * 64 + mi * 16 + lh * 8 + r;
                const float v = a[r] + bv;
                if (OUT_BF16)
                    ((unsigned short*)Cout)[(size_t)row * N + col] = f2bf(v);
                else
                    ((float*)Cout)[(size_t)row * N + col] = v;
            }
        }
    }
}

// ---------------------------------------------------------------------------
// Flash attention (causal). Grid: (T/64, H, B), 128 threads = 4 waves.
// Each wave owns a 16-row query tile; key tiles of 32 staged in LDS
// (K async row-major, V transposed for contiguous fragment loads),
// double-buffered across the key loop.
// qkv: [B*T, 3E] bf16 (Q at col 0, K at col E, V at col 2E, head h at h*64)
// y:   [B*T, E]  bf16
// ---------------------------------------------------------------------------
__global__ __launch_bounds__(128) void flash_attn(
    const unsigned short* __restrict__ qkv,
    unsigned short* __restrict__ y, int T) {
    constexpr int E3 = 3072, E = 1024, HD = 64;
    constexpr int LDK  = 72;   // Ks: [32 keys][LDK], 144B rows (16B aligned)
    constexpr int LDVT = 48;   // Vt: [64 dims][LDVT], key-minor, 96B rows
    __shared__ unsigned short Ks[2][32 * LDK];
    __shared__ unsigned short Vt[2][64 * LDVT];
    __shared__ unsigned short Ps[4][16 * 32];

    const int qb   = blockIdx.x;
    const int h    = blockIdx.y;
    const int bb   = blockIdx.z;
    const int wave = threadIdx.x >> 5;
    const int lane = threadIdx.x & 31;
    const int ln = lane & 15, lh = lane >> 4;

    const size_t rowbase = (size_t)bb * T;
    const int q0 = qb * 64;
    const int qw = q0 + wave * 16;

    // Q fragments straight from global (A layout, row stride 3E)
    const unsigned short* qptr = qkv + (rowbase + qw) * (size_t)E3 + h * HD;
    const v16bf qa0 = load_frag_A(qptr, E3);       // d = 0..31
    const v16bf qa1 = load_frag_A(qptr + 32, E3);  // d = 32..63

    float mrun[8], lrun[8];
    v8f oacc[4];
#pragma unroll
    for (int r = 0; r < 8; ++r) { mrun[r] = -1e30f; lrun[r] = 0.f; }
#pragma unroll
    for (int i = 0; i < 4; ++i) oacc[i] = vzero8();

    const int fr = threadIdx.x >> 2;            // 0..31: key row of tile
    const int fc = (threadIdx.x & 3) * 16;      // dim chunk (16 elems = 32B)
    const int nkt = (q0 + 64) / 32;             // causal: 2*qb + 2 key tiles

    // ---- prologue: stage key-tile 0 ----
    copy64B_g2l(qkv + (rowbase + fr) * (size_t)E3 + E + h * HD + fc,
                &Ks[0][fr * LDK + fc]);
    {
        Pack16 p;
        const uint4* vsrc = (const uint4*)(qkv + (rowbase + fr) * (size_t)E3 +
                                           2 * E + h * HD + fc);
        p.v[0] = vsrc[0]; p.v[1] = vsrc[1];
#pragma unroll
        for (int i = 0; i < 16; ++i) Vt[0][(fc + i) * LDVT + fr] = p.h[i];
    }
    wait_async();
    __syncthreads();

    int cur = 0;
    for (int kt = 0; kt < nkt; ++kt) {
        const bool has_next = (kt + 1) < nkt;
        Pack16 nv;
        if (has_next) {
            const size_t trow = (rowbase + (kt + 1) * 32 + fr) * (size_t)E3;
            copy64B_g2l(qkv + trow + E + h * HD + fc,
                        &Ks[cur ^ 1][fr * LDK + fc]);
            const uint4* vsrc = (const uint4*)(qkv + trow + 2 * E + h * HD + fc);
            nv.v[0] = vsrc[0]; nv.v[1] = vsrc[1];
        }

        // ---- S = Q @ K^T : load all 4 K fragments, then 4 WMMAs ----
        const unsigned short* kb = &Ks[cur][0];
        v16bf kf00 = load_frag_BT(kb, LDK);                 // keys 0-15, d 0-31
        v16bf kf01 = load_frag_BT(kb + 32, LDK);            // keys 0-15, d 32-63
        v16bf kf10 = load_frag_BT(kb + 16 * LDK, LDK);      // keys 16-31, d 0-31
        v16bf kf11 = load_frag_BT(kb + 16 * LDK + 32, LDK); // keys 16-31, d 32-63
        v8f s0 = vzero8(), s1 = vzero8();
        s0 = WMMA_BF16(qa0, kf00, s0);
        s1 = WMMA_BF16(qa0, kf10, s1);
        s0 = WMMA_BF16(qa1, kf01, s0);
        s1 = WMMA_BF16(qa1, kf11, s1);

        // ---- online softmax (f32) ----
        const float scale = 0.125f;   // 1/sqrt(64)
        const int keybase = kt * 32;
#pragma unroll
        for (int r = 0; r < 8; ++r) {
            const int qrow = qw + lh * 8 + r;
            float v0 = s0[r] * scale;
            float v1 = s1[r] * scale;
            if (keybase + ln > qrow)      v0 = -1e30f;
            if (keybase + 16 + ln > qrow) v1 = -1e30f;
            const float rm   = rowmax16(fmaxf(v0, v1));
            const float mnew = fmaxf(mrun[r], rm);
            const float alpha = __expf(mrun[r] - mnew);
            v0 = __expf(v0 - mnew);
            v1 = __expf(v1 - mnew);
            const float rs = rowsum16(v0 + v1);
            lrun[r] = lrun[r] * alpha + rs;
            mrun[r] = mnew;
#pragma unroll
            for (int i = 0; i < 4; ++i) oacc[i][r] *= alpha;
            const int m = lh * 8 + r;
            Ps[wave][m * 32 + ln]      = f2bf(v0);
            Ps[wave][m * 32 + 16 + ln] = f2bf(v1);
        }

        // ---- O += P @ V : load P + all 4 V fragments, then 4 WMMAs ----
        const v16bf pa = load_frag_A(&Ps[wave][0], 32);
        v16bf vf[4];
#pragma unroll
        for (int i = 0; i < 4; ++i)
            vf[i] = load_frag_BT(&Vt[cur][(i * 16) * LDVT], LDVT);
#pragma unroll
        for (int i = 0; i < 4; ++i)
            oacc[i] = WMMA_BF16(pa, vf[i], oacc[i]);

        if (has_next) {
#pragma unroll
            for (int i = 0; i < 16; ++i)
                Vt[cur ^ 1][(fc + i) * LDVT + fr] = nv.h[i];
        }
        wait_async();
        __syncthreads();
        cur ^= 1;
    }

    // normalize + store y (bf16, [B*T, E] with col = h*64 + d)
#pragma unroll
    for (int i = 0; i < 4; ++i) {
        const int d = h * HD + i * 16 + ln;
#pragma unroll
        for (int r = 0; r < 8; ++r) {
            const int t = qw + lh * 8 + r;
            const float o = oacc[i][r] / lrun[r];
            y[(rowbase + t) * (size_t)E + d] = f2bf(o);
        }
    }
}

// ---------------------------------------------------------------------------
// Host launcher
// ---------------------------------------------------------------------------
extern "C" void kernel_launch(void* const* d_in, const int* in_sizes, int n_in,
                              void* d_out, int out_size, void* d_ws, size_t ws_size,
                              hipStream_t stream) {
    constexpr int Bb = 4, T = 2048, E = 1024;
    constexpr int M  = Bb * T;          // 8192
    constexpr int E3 = 3 * E;           // 3072

    const float* x    = (const float*)d_in[0];
    const float* Wqkv = (const float*)d_in[1];
    const float* bqkv = (const float*)d_in[2];
    const float* Wout = (const float*)d_in[3];
    const float* bout = (const float*)d_in[4];
    float* out = (float*)d_out;

    unsigned short* xb    = (unsigned short*)d_ws;                   // [M,E]
    unsigned short* wqkvb = xb    + (size_t)M * E;                   // [E,3E]
    unsigned short* woutb = wqkvb + (size_t)E * E3;                  // [E,E]
    unsigned short* qkvb  = woutb + (size_t)E * E;                   // [M,3E]
    unsigned short* yb    = qkvb  + (size_t)M * E3;                  // [M,E]

    // 1) f32 -> bf16 staging
    cvt_f32_bf16<<<2048, 256, 0, stream>>>(x,    xb,    (M * E)  / 4);
    cvt_f32_bf16<<<1024, 256, 0, stream>>>(Wqkv, wqkvb, (E * E3) / 4);
    cvt_f32_bf16<<<512,  256, 0, stream>>>(Wout, woutb, (E * E)  / 4);

    // 2) qkv = x @ Wqkv + b_qkv   (bf16 out)
    gemm_bf16<true><<<dim3(E3 / 128, M / 128), 256, 0, stream>>>(
        xb, wqkvb, bqkv, qkvb, M, E3, E);

    // 3) flash attention -> y (bf16)
    flash_attn<<<dim3(T / 64, 16, Bb), 128, 0, stream>>>(qkvb, yb, T);

    // 4) out = y @ Wout + b_out   (f32 out)
    gemm_bf16<false><<<dim3(E / 128, M / 128), 256, 0, stream>>>(
        yb, woutb, bout, out, M, E, E);
}
